// LTFGW_log_90082644066817
// MI455X (gfx1250) — matrix operand order repaired and checked by reading, not compiled, not run on hardware
//
#include <hip/hip_runtime.h>
#include <hip/hip_bf16.h>

#define N_NODES 10000
#define DEG     16
#define M_LOC   17            // 1 + DEG
#define K_TPL   16
#define TPL_N   10
#define N_FEAT  128
#define N_SINK  10
#define N_CG    4

typedef __attribute__((ext_vector_type(16))) _Float16 v16h;
typedef __attribute__((ext_vector_type(8)))  float    v8f;
typedef __attribute__((ext_vector_type(2)))  float    v2f;

__device__ __forceinline__ float wave_sum(float v) {
#pragma unroll
  for (int m = 16; m >= 1; m >>= 1) v += __shfl_xor(v, m, 32);
  return v;
}
__device__ __forceinline__ float wave_max(float v) {
#pragma unroll
  for (int m = 16; m >= 1; m >>= 1) v = fmaxf(v, __shfl_xor(v, m, 32));
  return v;
}

// ---------------------------------------------------------------------------
// Kernel 1 (launched): G[N,160] = x[N,128] @ tf[160,128]^T, exact f32 via
// v_wmma_f32_16x16x4_f32.  One wave per 16x16 tile, K=128 in 32 steps of 4.
// A (16x4 f32, 2 VGPRs): lane l holds A[l&15][k0 + 2*(l>>4) + j], j=0,1.
// B (4x16 f32, 2 VGPRs): lane l holds B[k0 + 2*(l>>4) + j][l&15] (lane-major,
//   K split mirrors A per ISA 7.12.2 32-bit layouts).
// D (16x16 f32, 8 VGPRs): vgpr r -> row r + 8*(l>>4), col l&15.
// GEMM is memory-bound (~12 MB @ 23.3 TB/s >> 0.41 GFLOP), so full f32
// precision is free; no reason to downconvert.
// ---------------------------------------------------------------------------
__global__ __launch_bounds__(32) void feat_gemm_f32_kernel(
    const float* __restrict__ x, const float* __restrict__ tf,
    float* __restrict__ G)
{
  const int rt   = blockIdx.x;            // 0..624  (row tile)
  const int ct   = blockIdx.y;            // 0..9    (col tile)
  const int lane = threadIdx.x;
  const int hi   = lane >> 4;
  const int row  = rt * 16 + (lane & 15);
  const int col  = ct * 16 + (lane & 15);
  const float* xr = x  + (size_t)row * N_FEAT;
  const float* tr = tf + (size_t)col * N_FEAT;
  v8f acc = {};
#pragma unroll
  for (int k0 = 0; k0 < N_FEAT; k0 += 4) {
    v2f a, b;
    a[0] = xr[k0 + 2 * hi];
    a[1] = xr[k0 + 2 * hi + 1];
    b[0] = tr[k0 + 2 * hi];
    b[1] = tr[k0 + 2 * hi + 1];
    acc = __builtin_amdgcn_wmma_f32_16x16x4_f32(
        false, a, false, b, (short)0, acc, false, false);
  }
#pragma unroll
  for (int r = 0; r < 8; ++r) {
    const int m = r + 8 * hi;
    G[(size_t)(rt * 16 + m) * (K_TPL * TPL_N) + col] = acc[r];
  }
}

// ---------------------------------------------------------------------------
// Kernel 1b (kept compiled as known-good WMMA fallback shape; not launched):
// same GEMM via v_wmma_f32_16x16x32_f16.
// ---------------------------------------------------------------------------
__global__ __launch_bounds__(32) void feat_gemm_f16_kernel(
    const float* __restrict__ x, const float* __restrict__ tf,
    float* __restrict__ G)
{
  const int rt   = blockIdx.x;
  const int ct   = blockIdx.y;
  const int lane = threadIdx.x;
  const int rb = lane & 15, hi = lane >> 4;
  const int row = rt * 16 + rb;
  const int col = ct * 16 + (lane & 15);
  v8f acc = {};
#pragma unroll
  for (int kc = 0; kc < 4; ++kc) {
    const int k0 = kc * 32;
    v16h a, b;
    const float* xr = x + (size_t)row * N_FEAT + k0;
#pragma unroll
    for (int j = 0; j < 8; ++j) {
      const int kk = ((j < 4) ? 2 * j : 16 + 2 * (j - 4)) + hi * 8;
      a[2 * j]     = (_Float16)xr[kk];
      a[2 * j + 1] = (_Float16)xr[kk + 1];
    }
    const float* tr = tf + (size_t)col * N_FEAT + k0;
#pragma unroll
    for (int j = 0; j < 8; ++j) {
      const int kk = ((j < 4) ? 2 * j : 16 + 2 * (j - 4)) + hi * 8;
      b[2 * j]     = (_Float16)tr[kk];
      b[2 * j + 1] = (_Float16)tr[kk + 1];
    }
    acc = __builtin_amdgcn_wmma_f32_16x16x32_f16(
        false, a, false, b, (short)0, acc, false, false);
  }
#pragma unroll
  for (int r = 0; r < 8; ++r) {
    const int m = r + 8 * hi;
    G[(size_t)(rt * 16 + m) * (K_TPL * TPL_N) + col] = acc[r];
  }
}

// ---------------------------------------------------------------------------
// Kernel 2: squared norms  xx[i] = |x_i|^2 , ff[kb] = |tf_kb|^2
// ---------------------------------------------------------------------------
__global__ void norms_kernel(const float* __restrict__ x,
                             const float* __restrict__ tf,
                             float* __restrict__ xx, float* __restrict__ ff)
{
  const int t = blockIdx.x * blockDim.x + threadIdx.x;
  if (t < N_NODES) {
    const float* r = x + (size_t)t * N_FEAT;
    float s = 0.f;
    for (int j = 0; j < N_FEAT; ++j) s += r[j] * r[j];
    xx[t] = s;
  }
  if (t < K_TPL * TPL_N) {
    const float* r = tf + (size_t)t * N_FEAT;
    float s = 0.f;
    for (int j = 0; j < N_FEAT; ++j) s += r[j] * r[j];
    ff[t] = s;
  }
}

// ---------------------------------------------------------------------------
// Kernel 3: FGW conditional-gradient solver.
// One block (512 thr = 16 waves) per node; wave k handles template k.
// Lanes 0..16 each own one row of all [17,10] matrices in registers.
// The gathered G rows are pulled memory->LDS with ASYNCcnt-tracked
// global_load_async_to_lds_b128 (no VGPR bounce).
// ---------------------------------------------------------------------------
__global__ __launch_bounds__(512) void fgw_kernel(
    const float* __restrict__ xx, const float* __restrict__ ff,
    const float* __restrict__ G,  const float* __restrict__ tpl,
    const float* __restrict__ q0, const float* __restrict__ alpha0,
    const int*  __restrict__ dst, float* __restrict__ out)
{
  __shared__ int   s_local[M_LOC];
  __shared__ int   s_nb[M_LOC][DEG];
  __shared__ float s_A[M_LOC][M_LOC];
  __shared__ float s_C1[M_LOC][M_LOC + 1];
  __shared__ __align__(16) float s_G[M_LOC][K_TPL * TPL_N];  // 160 fl = 640 B rows
  __shared__ float s_xxl[M_LOC];
  __shared__ float s_C2[K_TPL][TPL_N][TPL_N];
  __shared__ float s_S[K_TPL][M_LOC][12];     // per-wave staging tile
  __shared__ float s_gv[K_TPL][TPL_N];

  const int i    = blockIdx.x;
  const int tid  = threadIdx.x;
  const int k    = tid >> 5;
  const int lane = tid & 31;
  const bool rowOK = lane < M_LOC;

  // ---- node-shared setup ----
  if (tid < M_LOC) s_local[tid] = (tid == 0) ? i : dst[(size_t)i * DEG + (tid - 1)];
  __syncthreads();

  // Warm L2 for the gathered rows, then async-copy them straight into LDS.
  if (tid < M_LOC)
    __builtin_prefetch(G + (size_t)s_local[tid] * (K_TPL * TPL_N), 0, 1);
  for (int t = tid; t < M_LOC * 40; t += 512) {        // 40 x 16B chunks per row
    const int a = t / 40, c4 = t % 40;
    const float* gp = G + (size_t)s_local[a] * (K_TPL * TPL_N) + c4 * 4;
    const unsigned lds = (unsigned)(size_t)&s_G[a][c4 * 4];
    asm volatile("global_load_async_to_lds_b128 %0, %1, off"
                 :: "v"(lds), "v"(gp) : "memory");
  }

  for (int t = tid; t < M_LOC * DEG; t += 512) {
    const int a = t / DEG, j = t % DEG;
    s_nb[a][j] = dst[(size_t)s_local[a] * DEG + j];
  }
  if (tid < M_LOC) s_xxl[tid] = xx[s_local[tid]];
  for (int t = tid; t < K_TPL * TPL_N * TPL_N; t += 512)
    ((float*)s_C2)[t] = tpl[t];

  asm volatile("s_wait_asynccnt 0x0" ::: "memory");    // s_G ready (this wave)
  __syncthreads();

  for (int t = tid; t < M_LOC * M_LOC; t += 512) {
    const int a = t / M_LOC, b = t % M_LOC;
    const int lb = s_local[b];
    float v = 0.f;
    for (int j = 0; j < DEG; ++j) v = fmaxf(v, (s_nb[a][j] == lb) ? 1.f : 0.f);
    s_A[a][b] = v;
  }
  __syncthreads();
  for (int t = tid; t < M_LOC * M_LOC; t += 512) {
    const int a = t / M_LOC, b = t % M_LOC;
    s_C1[a][b] = fmaxf(s_A[a][b], s_A[b][a]);
  }
  __syncthreads();

  // ---- per-wave scalars ----
  const float alpha = 1.f / (1.f + __expf(-alpha0[0]));
  const float p     = 1.f / (float)M_LOC;
  const float logp  = -__logf((float)M_LOC);

  // softmax(q0[k,:]) computed redundantly per lane
  float qv[TPL_N];
  float mxq = -1e30f;
#pragma unroll
  for (int b = 0; b < TPL_N; ++b) mxq = fmaxf(mxq, q0[k * TPL_N + b]);
  float sq = 0.f;
#pragma unroll
  for (int b = 0; b < TPL_N; ++b) { qv[b] = __expf(q0[k * TPL_N + b] - mxq); sq += qv[b]; }
  const float lsq = __logf(sq);
  const float isq = 1.f / sq;
#pragma unroll
  for (int b = 0; b < TPL_N; ++b) qv[b] *= isq;
  const float logq_own = (lane < TPL_N) ? (q0[k * TPL_N + lane] - mxq - lsq) : 0.f;

  // (C2*C2) @ q  (column vector, replicated per lane)
  float c2q[TPL_N];
#pragma unroll
  for (int b = 0; b < TPL_N; ++b) {
    float s = 0.f;
    for (int c = 0; c < TPL_N; ++c) { const float v = s_C2[k][b][c]; s += v * v * qv[c]; }
    c2q[b] = s;
  }
  // (C1*C1) @ p  (C1 binary -> p * rowsum)
  float c1p = 0.f;
  if (rowOK) { for (int c = 0; c < M_LOC; ++c) c1p += s_C1[lane][c]; c1p *= p; }

  // M row, T row
  float M[TPL_N], T[TPL_N];
#pragma unroll
  for (int b = 0; b < TPL_N; ++b) {
    M[b] = rowOK ? (s_xxl[lane] + ff[k * TPL_N + b] - 2.f * s_G[lane][k * TPL_N + b]) : 0.f;
    T[b] = rowOK ? (p * qv[b]) : 0.f;
  }

  float U[TPL_N], CTC[TPL_N];

  // ---- conditional gradient ----
  for (int it = 0; it < N_CG; ++it) {
    // CTC = C1 @ T @ C2
    if (rowOK) { for (int b = 0; b < TPL_N; ++b) s_S[k][lane][b] = T[b]; }
    __syncthreads();
#pragma unroll
    for (int b = 0; b < TPL_N; ++b) {
      float s = 0.f;
      if (rowOK) for (int c = 0; c < M_LOC; ++c) s += s_C1[lane][c] * s_S[k][c][b];
      U[b] = s;
    }
#pragma unroll
    for (int b = 0; b < TPL_N; ++b) {
      float s = 0.f;
      if (rowOK) for (int c = 0; c < TPL_N; ++c) s += U[c] * s_C2[k][c][b];
      CTC[b] = s;
    }
    __syncthreads();

    float Gr[TPL_N];
    float amax = 0.f;
#pragma unroll
    for (int b = 0; b < TPL_N; ++b) {
      const float g = rowOK ? ((1.f - alpha) * M[b] + 2.f * alpha * (c1p + c2q[b] - 2.f * CTC[b])) : 0.f;
      Gr[b] = g;
      amax = fmaxf(amax, fabsf(g));
    }
    amax = wave_max(amax);
    const float reg  = 0.01f * amax + 1e-8f;
    const float rinv = 1.f / reg;

    // log-domain Sinkhorn
    float f = 0.f, gvv[TPL_N];
#pragma unroll
    for (int b = 0; b < TPL_N; ++b) gvv[b] = 0.f;
    for (int s = 0; s < N_SINK; ++s) {
      if (rowOK) {
        float m = -1e30f;
#pragma unroll
        for (int b = 0; b < TPL_N; ++b) m = fmaxf(m, (gvv[b] - Gr[b]) * rinv);
        float ss = 0.f;
#pragma unroll
        for (int b = 0; b < TPL_N; ++b) ss += __expf((gvv[b] - Gr[b]) * rinv - m);
        f = reg * logp - reg * (m + __logf(ss));
#pragma unroll
        for (int b = 0; b < TPL_N; ++b) s_S[k][lane][b] = (f - Gr[b]) * rinv;
      }
      __syncthreads();
      if (lane < TPL_N) {
        float m = -1e30f;
        for (int c = 0; c < M_LOC; ++c) m = fmaxf(m, s_S[k][c][lane]);
        float ss = 0.f;
        for (int c = 0; c < M_LOC; ++c) ss += __expf(s_S[k][c][lane] - m);
        s_gv[k][lane] = reg * logq_own - reg * (m + __logf(ss));
      }
      __syncthreads();
#pragma unroll
      for (int b = 0; b < TPL_N; ++b) gvv[b] = s_gv[k][b];
    }

    // D = Tn - T ;  CDC = C1 @ D @ C2
    float D[TPL_N];
#pragma unroll
    for (int b = 0; b < TPL_N; ++b) {
      const float Tn = rowOK ? __expf((f + gvv[b] - Gr[b]) * rinv) : 0.f;
      D[b] = Tn - T[b];
    }
    if (rowOK) { for (int b = 0; b < TPL_N; ++b) s_S[k][lane][b] = D[b]; }
    __syncthreads();
    float UD[TPL_N], CDC[TPL_N];
#pragma unroll
    for (int b = 0; b < TPL_N; ++b) {
      float s = 0.f;
      if (rowOK) for (int c = 0; c < M_LOC; ++c) s += s_C1[lane][c] * s_S[k][c][b];
      UD[b] = s;
    }
#pragma unroll
    for (int b = 0; b < TPL_N; ++b) {
      float s = 0.f;
      if (rowOK) for (int c = 0; c < TPL_N; ++c) s += UD[c] * s_C2[k][c][b];
      CDC[b] = s;
    }
    __syncthreads();

    float r0 = 0.f, r1 = 0.f, r2 = 0.f, r3 = 0.f;
    if (rowOK) {
#pragma unroll
      for (int b = 0; b < TPL_N; ++b) {
        r0 += CDC[b] * D[b];
        r1 += M[b] * D[b];
        r2 += (c1p + c2q[b]) * D[b];
        r3 += CTC[b] * D[b];
      }
    }
    r0 = wave_sum(r0); r1 = wave_sum(r1); r2 = wave_sum(r2); r3 = wave_sum(r3);
    const float ac = -2.f * alpha * r0;
    const float bc = (1.f - alpha) * r1 + alpha * (r2 - 4.f * r3);
    const float tstep = (ac > 0.f)
        ? fminf(fmaxf(-bc / (2.f * ac + 1e-16f), 0.f), 1.f)
        : ((ac + bc < 0.f) ? 1.f : 0.f);
#pragma unroll
    for (int b = 0; b < TPL_N; ++b) T[b] += tstep * D[b];
  }

  // ---- final FGW value ----
  if (rowOK) { for (int b = 0; b < TPL_N; ++b) s_S[k][lane][b] = T[b]; }
  __syncthreads();
#pragma unroll
  for (int b = 0; b < TPL_N; ++b) {
    float s = 0.f;
    if (rowOK) for (int c = 0; c < M_LOC; ++c) s += s_C1[lane][c] * s_S[k][c][b];
    U[b] = s;
  }
#pragma unroll
  for (int b = 0; b < TPL_N; ++b) {
    float s = 0.f;
    if (rowOK) for (int c = 0; c < TPL_N; ++c) s += U[c] * s_C2[k][c][b];
    CTC[b] = s;
  }
  float d0 = 0.f, d1 = 0.f;
  if (rowOK) {
#pragma unroll
    for (int b = 0; b < TPL_N; ++b) {
      d0 += M[b] * T[b];
      d1 += (c1p + c2q[b] - 2.f * CTC[b]) * T[b];
    }
  }
  d0 = wave_sum(d0); d1 = wave_sum(d1);
  if (lane == 0)
    out[(size_t)i * K_TPL + k] = __logf((1.f - alpha) * d0 + alpha * d1);
}

// ---------------------------------------------------------------------------
extern "C" void kernel_launch(void* const* d_in, const int* in_sizes, int n_in,
                              void* d_out, int out_size, void* d_ws, size_t ws_size,
                              hipStream_t stream) {
  const float* x      = (const float*)d_in[0];
  const float* tpl    = (const float*)d_in[1];
  const float* tf     = (const float*)d_in[2];
  const float* q0     = (const float*)d_in[3];
  const float* alpha0 = (const float*)d_in[4];
  const int*   ei     = (const int*)d_in[5];
  const int*   dst    = ei + (size_t)N_NODES * DEG;   // edge_index row 1

  float* G  = (float*)d_ws;                           // [N, 160]
  float* xx = G + (size_t)N_NODES * K_TPL * TPL_N;    // [N]
  float* ff = xx + N_NODES;                           // [160]
  float* out = (float*)d_out;

  dim3 g1(N_NODES / 16, TPL_N);                       // 625 x 10 tiles
  feat_gemm_f32_kernel<<<g1, 32, 0, stream>>>(x, tf, G);
  norms_kernel<<<(N_NODES + 255) / 256, 256, 0, stream>>>(x, tf, xx, ff);
  fgw_kernel<<<N_NODES, 512, 0, stream>>>(xx, ff, G, tpl, q0, alpha0, dst, out);
}
